// udkws_aa2proc_29910152249987
// MI455X (gfx1250) — matrix operand order repaired and checked by reading, not compiled, not run on hardware
//
#include <hip/hip_runtime.h>
#include <hip/hip_bf16.h>
#include <math.h>

// ---------------------------------------------------------------------------
// MI455X (gfx1250) implementation. wave32, WMMA f32_16x16x32_f16 for all GEMMs.
// Sizes: B=512 T=100 F_IN=384 D_PROJ=128 D_M=100 NH=4 DH=25 D_FF=400 OUT=100.
// Rows flattened: M = B*T = 51200 = 3200 tiles of 16.
// Weights are pre-packed into WMMA B-fragment order (f16, zero-padded) so the
// hot GEMM loops do one contiguous 32B load per lane per fragment.
// GRU: input gates hoisted into a WMMA GEMM; recurrent weights LDS-resident.
// ---------------------------------------------------------------------------

typedef __attribute__((ext_vector_type(16))) _Float16 v16h;
typedef __attribute__((ext_vector_type(8)))  float    v8f;

#define DEVI __device__ __forceinline__

DEVI int laneid() { return (int)(threadIdx.x & 31u); }

DEVI v8f wmma_f16(v16h a, v16h b, v8f c) {
  // (neg_a, A, neg_b, B, c_mod, C, reuse_a, reuse_b)
  return __builtin_amdgcn_wmma_f32_16x16x32_f16(false, a, false, b, (short)0, c,
                                                false, false);
}

// A fragment (16x32 f16, M x K) from an LDS row-major tile.
// Lanes 0-15 row M=lane hold K = k0+0..15 (elems 0-15);
// lanes 16-31 hold K = k0+16..31 (matches B fragment order).
DEVI v16h a_frag_lds(const _Float16* A, int lda, int k0) {
  int lane = laneid();
  int row  = lane & 15;
  int kb   = k0 + ((lane & 16) ? 16 : 0);
  const _Float16* p = A + row * lda + kb;
  v16h a;
#pragma unroll
  for (int e = 0; e < 16; ++e) a[e] = p[e];
  return a;
}

// B fragment where element (k,n) is stored transposed in LDS: Bm[n*ldb + k].
// Contiguous 16-half read per lane -> 2x ds_load_b128.
DEVI v16h b_frag_lds_nmajor(const _Float16* Bm, int ldb, int k0, int n0) {
  int lane = laneid();
  int col  = (lane & 15) + n0;
  int kb   = k0 + ((lane & 16) ? 16 : 0);
  const _Float16* p = Bm + col * ldb + kb;
  v16h b;
#pragma unroll
  for (int e = 0; e < 16; ++e) b[e] = p[e];
  return b;
}

// B fragment from a pre-packed global weight buffer: fragments stored in VGPR
// order, one contiguous v16h (32B) per lane. NT = Npad/16 tiles.
DEVI v16h b_frag_packed(const _Float16* Wp, int NT, int kt, int nt) {
  const v16h* p = (const v16h*)Wp;
  return p[(size_t)(kt * NT + nt) * 32 + laneid()];
}

// ---------------------------------------------------------------------------
// K0: pack a f32 weight into WMMA B-fragment order, f16, zero-padded to
// Kpad (mult of 32) x Npad (mult of 16). transpose=0: B[k][n] = W[k*N+n]
// (W stored [K][N]); transpose=1: B[k][n] = W[n*K+k] (W stored [N][K]).
// Fragment (kt,nt): lane l, elem e -> k = kt*32 + (l&16?16:0) + e,
// n = nt*16 + (l&15).
// ---------------------------------------------------------------------------
__global__ void k_pack_w(const float* __restrict__ W, int K, int N,
                         int Kpad, int Npad, int transpose,
                         _Float16* __restrict__ out)
{
  int NT = Npad >> 4;
  int total = Kpad * Npad;
  for (int idx = blockIdx.x * blockDim.x + threadIdx.x; idx < total;
       idx += gridDim.x * blockDim.x) {
    int e    = idx & 15;
    int lane = (idx >> 4) & 31;
    int frag = idx >> 9;                 // 32 lanes * 16 elems = 512
    int nt   = frag % NT;
    int kt   = frag / NT;
    int col  = nt * 16 + (lane & 15);
    int k    = kt * 32 + ((lane & 16) ? 16 : 0) + e;
    float v = 0.f;
    if (k < K && col < N)
      v = transpose ? W[(size_t)col * K + k] : W[(size_t)k * N + col];
    out[idx] = (_Float16)v;
  }
}

// ---------------------------------------------------------------------------
// K1: fused LayerNorm(384) + GEMM(384->128) + bias + SiLU + row L2 norm.
// One block = 16 rows, 256 threads = 8 waves; wave w owns N-tile w (8x16=128).
// ---------------------------------------------------------------------------
__global__ void k_ln_proj(const float* __restrict__ X,   // [51200][384]
                          const float* __restrict__ g,   // [384]
                          const float* __restrict__ bta, // [384]
                          const _Float16* __restrict__ Wp, // packed [12][8] frags
                          const float* __restrict__ bias,// [128]
                          _Float16* __restrict__ out)    // [51200][128] f16
{
  __shared__ _Float16 As[16][384];
  __shared__ float    outS[16][128];
  __shared__ float    red[16][16], red2[16][16];

  int row0 = blockIdx.x * 16;
  int tid  = threadIdx.x;
  int r = tid >> 4, c = tid & 15;

  // ---- LayerNorm: 16 threads per row, 24 strided elements each ----
  float xr[24];
  float s = 0.f, sq = 0.f;
  const float* xrow = X + (size_t)(row0 + r) * 384;
#pragma unroll
  for (int i = 0; i < 24; ++i) {
    float v = xrow[c + i * 16];
    xr[i] = v; s += v; sq += v * v;
  }
  red[r][c] = s; red2[r][c] = sq;
  __syncthreads();
  if (c == 0) {
    float a = 0.f, b2 = 0.f;
    for (int i = 0; i < 16; ++i) { a += red[r][i]; b2 += red2[r][i]; }
    red[r][0] = a; red2[r][0] = b2;
  }
  __syncthreads();
  float mean = red[r][0] * (1.f / 384.f);
  float var  = red2[r][0] * (1.f / 384.f) - mean * mean;
  float rstd = rsqrtf(var + 1e-5f);
#pragma unroll
  for (int i = 0; i < 24; ++i) {
    int j = c + i * 16;
    As[r][j] = (_Float16)(((xr[i] - mean) * rstd) * g[j] + bta[j]);
  }
  __syncthreads();

  // ---- GEMM: K=384 (12 WMMA steps), N tile per wave ----
  int wave = tid >> 5;
  v8f acc = {};
  const v16h* wp = (const v16h*)Wp;
  for (int kt = 0; kt < 12; ++kt) {
    if (kt + 1 < 12)
      __builtin_prefetch((const void*)(wp + (size_t)((kt + 1) * 8 + wave) * 32 +
                                       laneid()), 0, 1);
    v16h a = a_frag_lds(&As[0][0], 384, kt * 32);
    v16h b = b_frag_packed(Wp, 8, kt, wave);
    acc = wmma_f16(a, b, acc);
  }
  {
    int lane = laneid();
    int col  = (lane & 15) + wave * 16;
    int rb   = (lane & 16) ? 8 : 0;
#pragma unroll
    for (int rr = 0; rr < 8; ++rr) {
      float v  = acc[rr] + bias[col];
      outS[rb + rr][col] = v / (1.f + __expf(-v));   // SiLU
    }
  }
  __syncthreads();

  // ---- row L2 normalization, write f16 ----
  float ps = 0.f;
#pragma unroll
  for (int i = 0; i < 8; ++i) { float v = outS[r][c + i * 16]; ps += v * v; }
  red[r][c] = ps;
  __syncthreads();
  if (c == 0) { float a = 0.f; for (int i = 0; i < 16; ++i) a += red[r][i]; red[r][0] = a; }
  __syncthreads();
  float scale = 1.f / fmaxf(sqrtf(red[r][0]), 1e-8f);
  _Float16* orow = out + (size_t)(row0 + r) * 128;
#pragma unroll
  for (int i = 0; i < 8; ++i) { int j = c + i * 16; orow[j] = (_Float16)(outS[r][j] * scale); }
}

// ---------------------------------------------------------------------------
// K2: sim[b] = an_n[b] (100x128) @ co_n[b]^T (128x100), masked. One block/batch.
// ---------------------------------------------------------------------------
__global__ void k_sim(const _Float16* __restrict__ an,
                      const _Float16* __restrict__ co,
                      const float* __restrict__ amask,
                      const float* __restrict__ cmask,
                      float* __restrict__ sim)          // [512][100][100]
{
  __shared__ _Float16 anS[112][128];
  __shared__ _Float16 coS[112][128];
  int b   = blockIdx.x;
  int tid = threadIdx.x;
  const _Float16* ap = an + (size_t)b * 100 * 128;
  const _Float16* cp = co + (size_t)b * 100 * 128;
  for (int i = tid; i < 112 * 128; i += 256) {
    int rr = i >> 7, cc = i & 127;
    _Float16 z = (_Float16)0.f;
    anS[rr][cc] = (rr < 100) ? ap[rr * 128 + cc] : z;
    coS[rr][cc] = (rr < 100) ? cp[rr * 128 + cc] : z;
  }
  __syncthreads();
  int wave = tid >> 5;
  float* simb = sim + (size_t)b * 100 * 100;
  for (int job = wave; job < 49; job += 8) {
    int ti = job / 7, tj = job % 7;
    v8f acc = {};
    for (int kt = 0; kt < 4; ++kt) {
      v16h a  = a_frag_lds(&anS[ti * 16][0], 128, kt * 32);
      v16h bb = b_frag_lds_nmajor(&coS[0][0], 128, kt * 32, tj * 16);
      acc = wmma_f16(a, bb, acc);
    }
    int lane = laneid();
    int col  = (lane & 15) + tj * 16;
    int rb   = (lane & 16) ? 8 : 0;
#pragma unroll
    for (int rr = 0; rr < 8; ++rr) {
      int i = ti * 16 + rb + rr;
      if (i < 100 && col < 100) {
        float m = amask[b * 100 + i] * cmask[b * 100 + col];
        simb[i * 100 + col] = acc[rr] * m;
      }
    }
  }
}

// ---------------------------------------------------------------------------
// K3: per-batch bilinear resize (la,lc derived from masks) + 5x5 Gaussian SAME.
// ---------------------------------------------------------------------------
__global__ void k_resize_conv(const float* __restrict__ sim,
                              const float* __restrict__ amask,
                              const float* __restrict__ cmask,
                              float* __restrict__ sm)   // [512][100][100]
{
  __shared__ float upS[100][104];
  __shared__ float r1[256], r2[256];
  int b   = blockIdx.x;
  int tid = threadIdx.x;

  float sa = 0.f, sc = 0.f;
  for (int i = tid; i < 100; i += 256) { sa += amask[b * 100 + i]; sc += cmask[b * 100 + i]; }
  r1[tid] = sa; r2[tid] = sc;
  __syncthreads();
  for (int s = 128; s > 0; s >>= 1) {
    if (tid < s) { r1[tid] += r1[tid + s]; r2[tid] += r2[tid + s]; }
    __syncthreads();
  }
  int la = (int)(r1[0] + 0.5f);
  int lc = (int)(r2[0] + 0.5f);

  const float* sb = sim + (size_t)b * 100 * 100;
  float fa = (float)(la - 1) / 99.f;
  float fc = (float)(lc - 1) / 99.f;
  for (int idx = tid; idx < 10000; idx += 256) {
    int rr = idx / 100, cc = idx % 100;
    float srr = rr * fa, scc = cc * fc;
    int r0 = (int)floorf(srr);
    int c0 = (int)floorf(scc);
    int r1i = min(r0 + 1, la - 1);
    int c1i = min(c0 + 1, lc - 1);
    float wr = srr - (float)r0, wc = scc - (float)c0;
    float v00 = sb[r0 * 100 + c0],  v01 = sb[r0 * 100 + c1i];
    float v10 = sb[r1i * 100 + c0], v11 = sb[r1i * 100 + c1i];
    float top = v00 * (1.f - wc) + v01 * wc;
    float bot = v10 * (1.f - wc) + v11 * wc;
    upS[rr][cc] = top * (1.f - wr) + bot * wr;
  }
  __syncthreads();

  float gk[5][5]; float gs = 0.f;
#pragma unroll
  for (int i = 0; i < 5; ++i)
#pragma unroll
    for (int j = 0; j < 5; ++j) {
      float dx = (float)i - 2.f, dy = (float)j - 2.f;
      float v = __expf(-(dx * dx + dy * dy) * 0.5f);
      gk[i][j] = v; gs += v;
    }
  float inv = 1.f / gs;
  float* smb = sm + (size_t)b * 100 * 100;
  for (int idx = tid; idx < 10000; idx += 256) {
    int rr = idx / 100, cc = idx % 100;
    float acc = 0.f;
#pragma unroll
    for (int di = 0; di < 5; ++di) {
      int ri = rr + di - 2;
      if (ri < 0 || ri >= 100) continue;
#pragma unroll
      for (int dj = 0; dj < 5; ++dj) {
        int cj = cc + dj - 2;
        if (cj < 0 || cj >= 100) continue;
        acc += gk[di][dj] * upS[ri][cj];
      }
    }
    smb[idx] = acc * inv;
  }
}

// ---------------------------------------------------------------------------
// K4: q/k/v = sm @ W{q,k,v} + b. 21 tile-jobs (3 mats x 7 N-tiles) over 8 waves.
// Packed weights: Kpad=128, Npad=112 (NT=7). Output f16, ld=112, zero-padded.
// ---------------------------------------------------------------------------
__global__ void k_qkv(const float* __restrict__ sm,
                      const _Float16* Wqp, const float* bq,
                      const _Float16* Wkp, const float* bk,
                      const _Float16* Wvp, const float* bv,
                      _Float16* q, _Float16* k, _Float16* v)
{
  __shared__ _Float16 Xs[16][128];
  int row0 = blockIdx.x * 16;
  int tid  = threadIdx.x;
  const float* xp = sm + (size_t)row0 * 100;
  for (int i = tid; i < 16 * 128; i += 256) {
    int rr = i >> 7, cc = i & 127;
    Xs[rr][cc] = (cc < 100) ? (_Float16)xp[rr * 100 + cc] : (_Float16)0.f;
  }
  __syncthreads();
  int wave = tid >> 5;
  for (int job = wave; job < 21; job += 8) {
    int which = job / 7, tn = job % 7;
    const _Float16* Wp   = (which == 0) ? Wqp : (which == 1) ? Wkp : Wvp;
    const float*    bias = (which == 0) ? bq  : (which == 1) ? bk  : bv;
    _Float16*       dst  = (which == 0) ? q   : (which == 1) ? k   : v;
    v8f acc = {};
    for (int kt = 0; kt < 4; ++kt) {
      v16h a  = a_frag_lds(&Xs[0][0], 128, kt * 32);
      v16h bb = b_frag_packed(Wp, 7, kt, tn);
      acc = wmma_f16(a, bb, acc);
    }
    int lane = laneid();
    int col  = (lane & 15) + tn * 16;
    int rb   = (lane & 16) ? 8 : 0;
#pragma unroll
    for (int rr = 0; rr < 8; ++rr) {
      int gr = row0 + rb + rr;
      float val = (col < 100) ? (acc[rr] + bias[col]) : 0.f;
      dst[(size_t)gr * 112 + col] = (_Float16)val;
    }
  }
}

// ---------------------------------------------------------------------------
// K5: attention per (b,h): scores = q k^T / 5 (WMMA, K=25->32), softmax,
// ao = att @ v (WMMA, K=100->128). V staged TRANSPOSED (vT[d][t]) so both
// GEMMs use contiguous n-major LDS fragment reads. ao written into the (dead)
// q buffer: block (b,h) only touches its own disjoint 25-column head slice.
// ---------------------------------------------------------------------------
__global__ void k_attn(const _Float16* __restrict__ q,
                       const _Float16* __restrict__ k,
                       const _Float16* __restrict__ v,
                       _Float16* ao)
{
  __shared__ _Float16 qS[112][32];
  __shared__ _Float16 kS[112][32];
  __shared__ _Float16 vT[32][128];   // vT[d][t] = v[t][d], zero padded
  __shared__ _Float16 sS[112][128];
  int b   = blockIdx.x >> 2;
  int h   = blockIdx.x & 3;
  int tid = threadIdx.x;
  const size_t base = (size_t)b * 100 * 112 + (size_t)h * 25;

  for (int i = tid; i < 112 * 32; i += 256) {
    int rr = i >> 5, dd = i & 31;
    _Float16 zq = (_Float16)0.f, zk = zq;
    if (rr < 100 && dd < 25) {
      size_t off = base + (size_t)rr * 112 + dd;
      zq = q[off]; zk = k[off];
    }
    qS[rr][dd] = zq; kS[rr][dd] = zk;
  }
  for (int i = tid; i < 32 * 128; i += 256) {
    int dd = i >> 7, tt = i & 127;
    _Float16 zv = (_Float16)0.f;
    if (tt < 100 && dd < 25) zv = v[base + (size_t)tt * 112 + dd];
    vT[dd][tt] = zv;
  }
  __syncthreads();

  int wave = tid >> 5;
  for (int job = wave; job < 49; job += 8) {
    int ti = job / 7, tj = job % 7;
    v16h a  = a_frag_lds(&qS[ti * 16][0], 32, 0);
    v16h bb = b_frag_lds_nmajor(&kS[0][0], 32, 0, tj * 16);
    v8f acc = {};
    acc = wmma_f16(a, bb, acc);
    int lane = laneid();
    int col  = (lane & 15) + tj * 16;
    int rb   = (lane & 16) ? 8 : 0;
#pragma unroll
    for (int rr = 0; rr < 8; ++rr)
      sS[ti * 16 + rb + rr][col] = (_Float16)(acc[rr] * 0.2f);  // 1/sqrt(25)
  }
  __syncthreads();

  // softmax over j (valid cols 0..99); zero K-padding for the next GEMM
  for (int i = tid; i < 112; i += 256) {
    if (i < 100) {
      float mx = -1e30f;
      for (int j = 0; j < 100; ++j) mx = fmaxf(mx, (float)sS[i][j]);
      float sum = 0.f;
      for (int j = 0; j < 100; ++j) sum += __expf((float)sS[i][j] - mx);
      float inv = 1.f / sum;
      for (int j = 0; j < 100; ++j)
        sS[i][j] = (_Float16)(__expf((float)sS[i][j] - mx) * inv);
      for (int j = 100; j < 128; ++j) sS[i][j] = (_Float16)0.f;
    } else {
      for (int j = 0; j < 128; ++j) sS[i][j] = (_Float16)0.f;
    }
  }
  __syncthreads();

  for (int job = wave; job < 14; job += 8) {
    int ti = job / 2, tn = job % 2;
    v8f acc = {};
    for (int kt = 0; kt < 4; ++kt) {
      v16h a  = a_frag_lds(&sS[ti * 16][0], 128, kt * 32);
      v16h bb = b_frag_lds_nmajor(&vT[0][0], 128, kt * 32, tn * 16);
      acc = wmma_f16(a, bb, acc);
    }
    int lane = laneid();
    int col  = (lane & 15) + tn * 16;
    int rb   = (lane & 16) ? 8 : 0;
#pragma unroll
    for (int rr = 0; rr < 8; ++rr) {
      int i = ti * 16 + rb + rr;
      if (i < 100 && col < 25)
        ao[base + (size_t)i * 112 + col] = (_Float16)acc[rr];
    }
  }
}

// ---------------------------------------------------------------------------
// K6: x = LayerNorm(sm + ao @ Wo + bo). One block = 16 rows. Wo packed (NT=7).
// ---------------------------------------------------------------------------
__global__ void k_attn_out_ln(const _Float16* __restrict__ ao,
                              const _Float16* __restrict__ Wop, const float* bo,
                              const float* __restrict__ sm,
                              const float* g1, const float* b1,
                              float* x)
{
  __shared__ _Float16 Aos[16][128];
  __shared__ float    outS[16][112];
  __shared__ float    red[16][17], red2[16][17];
  int row0 = blockIdx.x * 16;
  int tid  = threadIdx.x;
  for (int i = tid; i < 16 * 128; i += 256) {
    int rr = i >> 7, cc = i & 127;
    Aos[rr][cc] = (cc < 100) ? ao[(size_t)(row0 + rr) * 112 + cc] : (_Float16)0.f;
  }
  __syncthreads();
  int wave = tid >> 5;
  for (int job = wave; job < 7; job += 8) {
    v8f acc = {};
    for (int kt = 0; kt < 4; ++kt) {
      v16h a  = a_frag_lds(&Aos[0][0], 128, kt * 32);
      v16h bb = b_frag_packed(Wop, 7, kt, job);
      acc = wmma_f16(a, bb, acc);
    }
    int lane = laneid();
    int col  = (lane & 15) + job * 16;
    int rb   = (lane & 16) ? 8 : 0;
#pragma unroll
    for (int rr = 0; rr < 8; ++rr) {
      int gr = row0 + rb + rr;
      float val = 0.f;
      if (col < 100) val = acc[rr] + bo[col] + sm[(size_t)gr * 100 + col];
      outS[rb + rr][col] = val;
    }
  }
  __syncthreads();

  int r = tid >> 4, c = tid & 15;
  float s = 0.f, sq = 0.f;
  for (int j = c; j < 100; j += 16) { float vv = outS[r][j]; s += vv; sq += vv * vv; }
  red[r][c] = s; red2[r][c] = sq;
  __syncthreads();
  if (c == 0) {
    float a = 0.f, b2 = 0.f;
    for (int i = 0; i < 16; ++i) { a += red[r][i]; b2 += red2[r][i]; }
    red[r][16] = a; red2[r][16] = b2;
  }
  __syncthreads();
  float mean = red[r][16] * 0.01f;
  float var  = red2[r][16] * 0.01f - mean * mean;
  float rstd = rsqrtf(var + 1e-5f);
  for (int j = c; j < 100; j += 16)
    x[(size_t)(row0 + r) * 100 + j] = (outS[r][j] - mean) * rstd * g1[j] + b1[j];
}

// ---------------------------------------------------------------------------
// K7: x = LayerNorm(x + relu(x@Wff1+b)@Wff2+b). In-place safe per 16-row block.
// W1 packed (Kpad=128, NT=25), W2 packed (Kpad=416, NT=7).
// ---------------------------------------------------------------------------
__global__ void k_ffn_ln(const float* xin,
                         const _Float16* W1p, const float* bf1,
                         const _Float16* W2p, const float* bf2,
                         const float* g2, const float* b2v,
                         float* x)
{
  __shared__ _Float16 Xs[16][128];
  __shared__ _Float16 Hs[16][416];
  __shared__ float    outS[16][112];
  __shared__ float    red[16][17], red2[16][17];
  int row0 = blockIdx.x * 16;
  int tid  = threadIdx.x;
  for (int i = tid; i < 16 * 128; i += 256) {
    int rr = i >> 7, cc = i & 127;
    Xs[rr][cc] = (cc < 100) ? (_Float16)xin[(size_t)(row0 + rr) * 100 + cc]
                            : (_Float16)0.f;
  }
  __syncthreads();
  int wave = tid >> 5;
  for (int job = wave; job < 25; job += 8) {        // N = 400 = 25 tiles
    v8f acc = {};
    for (int kt = 0; kt < 4; ++kt) {
      v16h a  = a_frag_lds(&Xs[0][0], 128, kt * 32);
      v16h bb = b_frag_packed(W1p, 25, kt, job);
      acc = wmma_f16(a, bb, acc);
    }
    int lane = laneid();
    int col  = (lane & 15) + job * 16;
    int rb   = (lane & 16) ? 8 : 0;
#pragma unroll
    for (int rr = 0; rr < 8; ++rr)
      Hs[rb + rr][col] = (_Float16)fmaxf(acc[rr] + bf1[col], 0.f);  // ReLU
  }
  for (int i = tid; i < 16 * 16; i += 256)          // zero K-pad 400..415
    Hs[i >> 4][400 + (i & 15)] = (_Float16)0.f;
  __syncthreads();

  for (int job = wave; job < 7; job += 8) {
    v8f acc = {};
    for (int kt = 0; kt < 13; ++kt) {               // K = 416
      v16h a  = a_frag_lds(&Hs[0][0], 416, kt * 32);
      v16h bb = b_frag_packed(W2p, 7, kt, job);
      acc = wmma_f16(a, bb, acc);
    }
    int lane = laneid();
    int col  = (lane & 15) + job * 16;
    int rb   = (lane & 16) ? 8 : 0;
#pragma unroll
    for (int rr = 0; rr < 8; ++rr) {
      int gr = row0 + rb + rr;
      float val = 0.f;
      if (col < 100) val = acc[rr] + bf2[col] + xin[(size_t)gr * 100 + col];
      outS[rb + rr][col] = val;
    }
  }
  __syncthreads();

  int r = tid >> 4, c = tid & 15;
  float s = 0.f, sq = 0.f;
  for (int j = c; j < 100; j += 16) { float vv = outS[r][j]; s += vv; sq += vv * vv; }
  red[r][c] = s; red2[r][c] = sq;
  __syncthreads();
  if (c == 0) {
    float a = 0.f, b2 = 0.f;
    for (int i = 0; i < 16; ++i) { a += red[r][i]; b2 += red2[r][i]; }
    red[r][16] = a; red2[r][16] = b2;
  }
  __syncthreads();
  float mean = red[r][16] * 0.01f;
  float var  = red2[r][16] * 0.01f - mean * mean;
  float rstd = rsqrtf(var + 1e-5f);
  for (int j = c; j < 100; j += 16)
    x[(size_t)(row0 + r) * 100 + j] = (outS[r][j] - mean) * rstd * g2[j] + b2v[j];
}

// ---------------------------------------------------------------------------
// K8a: gi = X @ W_ih^T + b_ih for all (b,t) rows, via WMMA.
// W_ih^T packed with transpose=1: Kpad=128, Npad=304 (NT=19). gi ld = 304.
// ---------------------------------------------------------------------------
__global__ void k_gi(const float* __restrict__ x,       // [51200][100]
                     const _Float16* __restrict__ Wihp, // packed, NT=19
                     const float* __restrict__ bih,     // [300]
                     float* __restrict__ gi)            // [51200][304]
{
  __shared__ _Float16 Xs[16][128];
  int row0 = blockIdx.x * 16;
  int tid  = threadIdx.x;
  for (int i = tid; i < 16 * 128; i += 256) {
    int rr = i >> 7, cc = i & 127;
    Xs[rr][cc] = (cc < 100) ? (_Float16)x[(size_t)(row0 + rr) * 100 + cc]
                            : (_Float16)0.f;
  }
  __syncthreads();
  int wave = tid >> 5;
  for (int job = wave; job < 19; job += 8) {
    v8f acc = {};
    for (int kt = 0; kt < 4; ++kt) {
      v16h a  = a_frag_lds(&Xs[0][0], 128, kt * 32);
      v16h bb = b_frag_packed(Wihp, 19, kt, job);
      acc = wmma_f16(a, bb, acc);
    }
    int lane = laneid();
    int col  = (lane & 15) + job * 16;
    int rb   = (lane & 16) ? 8 : 0;
#pragma unroll
    for (int rr = 0; rr < 8; ++rr) {
      int gr = row0 + rb + rr;
      if (col < 300)
        gi[(size_t)gr * 304 + col] = acc[rr] + bih[col];
    }
  }
}

// ---------------------------------------------------------------------------
// K8b: GRU recurrence. One block per batch element, 320 threads, 100 steps.
// W_hh staged once into LDS as f16 (300x100 = 60 KB) -> per-step traffic is
// LDS-only plus one 300-float gi row. gi already contains b_ih.
// ---------------------------------------------------------------------------
__global__ void __launch_bounds__(320)
k_gru(const float* __restrict__ gi,    // [51200][304]
      const float* __restrict__ Whh,   // [300][100] f32
      const float* __restrict__ bhh,   // [300]
      float* __restrict__ hT)          // [512][100]
{
  __shared__ _Float16 WhS[300 * 100];
  __shared__ float hS[100];
  __shared__ float ghS[300];
  int b   = blockIdx.x;
  int tid = threadIdx.x;
  for (int i = tid; i < 30000; i += 320) WhS[i] = (_Float16)Whh[i];
  if (tid < 100) hS[tid] = 0.f;
  __syncthreads();

  for (int t = 0; t < 100; ++t) {
    if (tid < 300) {
      float gh = bhh[tid];
      const _Float16* wr = &WhS[tid * 100];
#pragma unroll 4
      for (int kk = 0; kk < 100; ++kk) gh += hS[kk] * (float)wr[kk];
      ghS[tid] = gh;
    }
    __syncthreads();
    float hn = 0.f;
    bool act = (tid < 100);
    if (act) {
      const float* gir = gi + ((size_t)b * 100 + t) * 304;
      float r = 1.f / (1.f + __expf(-(gir[tid] + ghS[tid])));
      float z = 1.f / (1.f + __expf(-(gir[100 + tid] + ghS[100 + tid])));
      float n = tanhf(gir[200 + tid] + r * ghS[200 + tid]);
      hn = (1.f - z) * n + z * hS[tid];
    }
    __syncthreads();
    if (act) hS[tid] = hn;
    __syncthreads();
  }
  if (tid < 100) hT[(size_t)b * 100 + tid] = hS[tid];
}

// ---------------------------------------------------------------------------
// K9: out[b] = (hT[b] @ W_fc + b_fc) @ W_dense + b_dense.
// ---------------------------------------------------------------------------
__global__ void k_head(const float* __restrict__ hT,
                       const float* __restrict__ Wfc, const float* __restrict__ bfc,
                       const float* __restrict__ Wd,  const float* __restrict__ bd,
                       float* __restrict__ out)
{
  __shared__ float red[128];
  int b   = blockIdx.x;
  int tid = threadIdx.x;   // 128
  float p = 0.f;
  if (tid < 100) {
    float f = bfc[tid];
    const float* h = hT + (size_t)b * 100;
    for (int kk = 0; kk < 100; ++kk) f += h[kk] * Wfc[kk * 100 + tid];
    p = f * Wd[tid];
  }
  red[tid] = p;
  __syncthreads();
  for (int s = 64; s > 0; s >>= 1) {
    if (tid < s) red[tid] += red[tid + s];
    __syncthreads();
  }
  if (tid == 0) out[b] = red[0] + bd[0];
}

// ---------------------------------------------------------------------------
extern "C" void kernel_launch(void* const* d_in, const int* in_sizes, int n_in,
                              void* d_out, int out_size, void* d_ws, size_t ws_size,
                              hipStream_t stream)
{
  (void)in_sizes; (void)n_in; (void)out_size; (void)ws_size;

  const float* anco  = (const float*)d_in[0];
  const float* amask = (const float*)d_in[1];
  const float* com   = (const float*)d_in[2];
  const float* cmask = (const float*)d_in[3];
  // d_in[4] anco_text, d_in[5] label: unused by the forward pass
  const float* ln_an_g  = (const float*)d_in[6];
  const float* ln_an_b  = (const float*)d_in[7];
  const float* W_an     = (const float*)d_in[8];
  const float* b_an     = (const float*)d_in[9];
  const float* ln_com_g = (const float*)d_in[10];
  const float* ln_com_b = (const float*)d_in[11];
  const float* W_com    = (const float*)d_in[12];
  const float* b_com    = (const float*)d_in[13];
  const float* Wq = (const float*)d_in[14]; const float* bq = (const float*)d_in[15];
  const float* Wk = (const float*)d_in[16]; const float* bk = (const float*)d_in[17];
  const float* Wv = (const float*)d_in[18]; const float* bv = (const float*)d_in[19];
  const float* Wo = (const float*)d_in[20]; const float* bo = (const float*)d_in[21];
  const float* ln1_g = (const float*)d_in[22]; const float* ln1_b = (const float*)d_in[23];
  const float* Wff1  = (const float*)d_in[24]; const float* bff1  = (const float*)d_in[25];
  const float* Wff2  = (const float*)d_in[26]; const float* bff2  = (const float*)d_in[27];
  const float* ln2_g = (const float*)d_in[28]; const float* ln2_b = (const float*)d_in[29];
  const float* W_ih  = (const float*)d_in[30]; const float* W_hh  = (const float*)d_in[31];
  const float* b_ih  = (const float*)d_in[32]; const float* b_hh  = (const float*)d_in[33];
  const float* W_fc  = (const float*)d_in[34]; const float* b_fc  = (const float*)d_in[35];
  const float* W_dns = (const float*)d_in[36]; const float* b_dns = (const float*)d_in[37];

  // Workspace carve-out (~175 MB total)
  char* ws = (char*)d_ws;
  size_t off = 0;
  auto carve = [&](size_t bytes) -> void* {
    void* p = ws + off;
    off = (off + bytes + 255) & ~(size_t)255;
    return p;
  };
  const size_t M = 512ull * 100ull;                      // 51200 rows
  _Float16* an_n = (_Float16*)carve(M * 128 * sizeof(_Float16));
  _Float16* co_n = (_Float16*)carve(M * 128 * sizeof(_Float16));
  float*    simb = (float*)   carve(M * 100 * sizeof(float));
  float*    smb  = (float*)   carve(M * 100 * sizeof(float));
  _Float16* qb   = (_Float16*)carve(M * 112 * sizeof(_Float16));
  _Float16* kb   = (_Float16*)carve(M * 112 * sizeof(_Float16));
  _Float16* vb   = (_Float16*)carve(M * 112 * sizeof(_Float16));
  float*    xb   = (float*)   carve(M * 100 * sizeof(float));
  float*    gib  = (float*)   carve(M * 304 * sizeof(float));
  float*    hb   = (float*)   carve(512ull * 100 * sizeof(float));
  // packed weights (f16, fragment order)
  _Float16* Wanp = (_Float16*)carve(384ull * 128 * sizeof(_Float16));
  _Float16* Wcop = (_Float16*)carve(384ull * 128 * sizeof(_Float16));
  _Float16* Wqp  = (_Float16*)carve(128ull * 112 * sizeof(_Float16));
  _Float16* Wkp  = (_Float16*)carve(128ull * 112 * sizeof(_Float16));
  _Float16* Wvp  = (_Float16*)carve(128ull * 112 * sizeof(_Float16));
  _Float16* Wop  = (_Float16*)carve(128ull * 112 * sizeof(_Float16));
  _Float16* W1p  = (_Float16*)carve(128ull * 400 * sizeof(_Float16));
  _Float16* W2p  = (_Float16*)carve(416ull * 112 * sizeof(_Float16));
  _Float16* Wihp = (_Float16*)carve(128ull * 304 * sizeof(_Float16));

  // 0) pack weights into WMMA B-fragment order
  k_pack_w<<<96, 256, 0, stream>>>(W_an, 384, 128, 384, 128, 0, Wanp);
  k_pack_w<<<96, 256, 0, stream>>>(W_com, 384, 128, 384, 128, 0, Wcop);
  k_pack_w<<<28, 256, 0, stream>>>(Wq, 100, 100, 128, 112, 0, Wqp);
  k_pack_w<<<28, 256, 0, stream>>>(Wk, 100, 100, 128, 112, 0, Wkp);
  k_pack_w<<<28, 256, 0, stream>>>(Wv, 100, 100, 128, 112, 0, Wvp);
  k_pack_w<<<28, 256, 0, stream>>>(Wo, 100, 100, 128, 112, 0, Wop);
  k_pack_w<<<100, 256, 0, stream>>>(Wff1, 100, 400, 128, 400, 0, W1p);
  k_pack_w<<<91, 256, 0, stream>>>(Wff2, 400, 100, 416, 112, 0, W2p);
  k_pack_w<<<76, 256, 0, stream>>>(W_ih, 100, 300, 128, 304, 1, Wihp); // B=W_ih^T

  // 1) LN + projection + SiLU + L2-norm (anchor, comparison)
  k_ln_proj<<<3200, 256, 0, stream>>>(anco, ln_an_g, ln_an_b, Wanp, b_an, an_n);
  k_ln_proj<<<3200, 256, 0, stream>>>(com, ln_com_g, ln_com_b, Wcop, b_com, co_n);
  // 2) masked cosine-similarity matrices
  k_sim<<<512, 256, 0, stream>>>(an_n, co_n, amask, cmask, simb);
  // 3) bilinear resize + gaussian smoothing
  k_resize_conv<<<512, 256, 0, stream>>>(simb, amask, cmask, smb);
  // 4) QKV projections
  k_qkv<<<3200, 256, 0, stream>>>(smb, Wqp, bq, Wkp, bk, Wvp, bv, qb, kb, vb);
  // 5) attention (ao overwrites the q buffer: disjoint head columns)
  k_attn<<<512 * 4, 256, 0, stream>>>(qb, kb, vb, qb);
  // 6) output projection + residual + LN1
  k_attn_out_ln<<<3200, 256, 0, stream>>>(qb, Wop, bo, smb, ln1_g, ln1_b, xb);
  // 7) FFN + residual + LN2 (in-place on xb)
  k_ffn_ln<<<3200, 256, 0, stream>>>(xb, W1p, bff1, W2p, bff2, ln2_g, ln2_b, xb);
  // 8) GRU: input gates via WMMA GEMM, then LDS-resident recurrence
  k_gi<<<3200, 256, 0, stream>>>(xb, Wihp, b_ih, gib);
  k_gru<<<512, 320, 0, stream>>>(gib, W_hh, b_hh, hb);
  // 9) head
  k_head<<<512, 128, 0, stream>>>(hb, W_fc, b_fc, W_dns, b_dns, (float*)d_out);
}
